// BasicAttentionLayer_21010980012407
// MI455X (gfx1250) — matrix operand order repaired and checked by reading, not compiled, not run on hardware
//
#include <hip/hip_runtime.h>

// ---------------------------------------------------------------------------
// BERT attention layer for MI455X (gfx1250, wave32, WMMA bf16 16x16x32).
//   K0a: convert from_tensor / to_tensor f32 -> bf16
//   K0b: transpose+convert Wq/Wk/Wv [K,N] f32 -> [N,K] bf16
//   K1 : fused QKV GEMM (bf16 WMMA, f32 acc, bias add), weight tiles staged in
//        LDS via async global->LDS copies (ASYNCcnt), double-buffered.
//   K2 : scores = QK^T/8 + mask adder, softmax fused in LDS, probs f32 -> d_out
//   K3 : ctx = P.V (P read f32 from d_out, cvt to bf16 fragments), f32 -> d_out
// ---------------------------------------------------------------------------

typedef __attribute__((ext_vector_type(16))) __bf16 v16bf;
typedef __attribute__((ext_vector_type(8)))  __bf16 v8bf;
typedef __attribute__((ext_vector_type(4)))  __bf16 v4bf;
typedef __attribute__((ext_vector_type(8)))  float  v8f;

// builtin parameter type for async LDS copies: int4 (gcc vector) pointers
typedef int v4i_vs __attribute__((vector_size(16)));
typedef __attribute__((address_space(1))) v4i_vs* as1_v4i_ptr;
typedef __attribute__((address_space(3))) v4i_vs* as3_v4i_ptr;

#define HIDDEN 1024
#define NHEAD  16
#define DH     64
#define BATCH  4
#define SEQ    1024
#define BH     (BATCH*NHEAD)   // 64

// ---- CDNA5 async global->LDS copy (ASYNCcnt) with safe fallback ------------
#if __has_builtin(__builtin_amdgcn_global_load_async_to_lds_b128)
#define HAVE_ASYNC_LDS 1
#else
#define HAVE_ASYNC_LDS 0
#endif

#if __has_builtin(__builtin_amdgcn_s_wait_asynccnt)
#define WAIT_ASYNC(n) __builtin_amdgcn_s_wait_asynccnt(n)
#elif HAVE_ASYNC_LDS
#define WAIT_ASYNC(n) asm volatile("s_wait_asynccnt %0" ::"n"(n) : "memory")
#else
#define WAIT_ASYNC(n) ((void)0)
#endif

// copy 16 bytes global -> LDS, per lane
static __device__ inline void async_copy16(const __bf16* g, __bf16* l) {
#if HAVE_ASYNC_LDS
  __builtin_amdgcn_global_load_async_to_lds_b128(
      (as1_v4i_ptr)g, (as3_v4i_ptr)l, 0, 0);
#else
  *(v8bf*)l = *(const v8bf*)g;
#endif
}

static __device__ inline v8f wmma_bf16(v16bf a, v16bf b, v8f c) {
  // (neg_a, A, neg_b, B, c_mod, C, reuse_a, reuse_b)
  return __builtin_amdgcn_wmma_f32_16x16x32_bf16(false, a, false, b, (short)0, c,
                                                 false, false);
}

// A fragment (16x32 bf16, row-major source, row stride ld elements).
static __device__ inline v16bf load_a_bf16(const __bf16* base, int ld, int lane) {
  int m = lane & 15, half = lane >> 4;
  const __bf16* p = base + m * ld + half * 8;
  v8bf lo = *(const v8bf*)(p);
  v8bf hi = *(const v8bf*)(p + 16);
  v16bf a;
#pragma unroll
  for (int i = 0; i < 8; i++) { a[i] = lo[i]; a[8 + i] = hi[i]; }
  return a;
}

// Same fragment but from an f32 source (probs), converted on the fly.
static __device__ inline v16bf load_a_f32(const float* base, int ld, int lane) {
  int m = lane & 15, half = lane >> 4;
  const float* p = base + m * ld + half * 8;
  float4 l0 = *(const float4*)(p);
  float4 l1 = *(const float4*)(p + 4);
  float4 h0 = *(const float4*)(p + 16);
  float4 h1 = *(const float4*)(p + 20);
  v16bf a;
  a[0] = (__bf16)l0.x;  a[1] = (__bf16)l0.y;  a[2]  = (__bf16)l0.z;  a[3]  = (__bf16)l0.w;
  a[4] = (__bf16)l1.x;  a[5] = (__bf16)l1.y;  a[6]  = (__bf16)l1.z;  a[7]  = (__bf16)l1.w;
  a[8] = (__bf16)h0.x;  a[9] = (__bf16)h0.y;  a[10] = (__bf16)h0.z;  a[11] = (__bf16)h0.w;
  a[12] = (__bf16)h1.x; a[13] = (__bf16)h1.y; a[14] = (__bf16)h1.z; a[15] = (__bf16)h1.w;
  return a;
}

// B fragment (32x16): source stored "n-major", element (k,n) at base[n*ld + k].
static __device__ inline v16bf load_b_bf16(const __bf16* base, int ld, int lane) {
  int n = lane & 15, half = lane >> 4;
  const __bf16* p = base + n * ld + half * 8;
  v8bf lo = *(const v8bf*)(p);
  v8bf hi = *(const v8bf*)(p + 16);
  v16bf b;
#pragma unroll
  for (int i = 0; i < 8; i++) { b[i] = lo[i]; b[8 + i] = hi[i]; }
  return b;
}

// ---------------------------------------------------------------------------
// K0a: f32 -> bf16 elementwise convert (vectorized).
__global__ __launch_bounds__(256) void cvt_kernel(const float* __restrict__ in,
                                                  __bf16* __restrict__ out) {
  int i = (blockIdx.x * 256 + threadIdx.x) * 4;
  float4 v = *(const float4*)(in + i);
  v4bf o;
  o[0] = (__bf16)v.x; o[1] = (__bf16)v.y; o[2] = (__bf16)v.z; o[3] = (__bf16)v.w;
  *(v4bf*)(out + i) = o;
}

// K0b: W [K=1024, N=1024] f32 -> Wt [N, K] bf16 (LDS-tiled transpose).
__global__ __launch_bounds__(256) void transpose_cvt_kernel(const float* __restrict__ W,
                                                            __bf16* __restrict__ Wt) {
  __shared__ float tile[32][33];
  int bx = blockIdx.x * 32, by = blockIdx.y * 32;
  int tx = threadIdx.x & 31, ty = threadIdx.x >> 5;  // 32 x 8
#pragma unroll
  for (int i = 0; i < 4; i++) {
    int k = by + ty + i * 8;
    tile[ty + i * 8][tx] = W[k * HIDDEN + bx + tx];
  }
  __syncthreads();
#pragma unroll
  for (int i = 0; i < 4; i++) {
    int n = bx + ty + i * 8;
    Wt[n * HIDDEN + by + tx] = (__bf16)tile[tx][ty + i * 8];
  }
}

// ---------------------------------------------------------------------------
// K1: QKV GEMM. grid = (16 heads, 32 row-tiles, 3 {Q,K,V}), block = 256 (8 waves).
// Each wave: 16 rows x 64 cols (4 accumulators). Weight tile (64x32 bf16, 4KB)
// staged in LDS via async global->LDS copies, double-buffered; A fragments are
// software-pipelined one iteration ahead. All four B fragments are loaded into
// distinct registers before the WMMA burst (single DScnt wait per k-step).
__global__ __launch_bounds__(256) void qkv_gemm_kernel(
    const __bf16* __restrict__ Xf, const __bf16* __restrict__ Xt,
    const __bf16* __restrict__ Wqt, const __bf16* __restrict__ Wkt,
    const __bf16* __restrict__ Wvt,
    const float* __restrict__ bq, const float* __restrict__ bk,
    const float* __restrict__ bv,
    __bf16* __restrict__ Qb, __bf16* __restrict__ Kb, __bf16* __restrict__ Vt) {
  int z = blockIdx.z;
  const __bf16* X   = (z == 0) ? Xf : Xt;
  const __bf16* W   = (z == 0) ? Wqt : (z == 1 ? Wkt : Wvt);
  const float* bias = (z == 0) ? bq : (z == 1 ? bk : bv);

  int h = blockIdx.x;        // one head per x-block (64 output columns)
  int n0 = h * 64;
  int wave = threadIdx.x >> 5, lane = threadIdx.x & 31;
  int rw = blockIdx.y * 128 + wave * 16;   // global row (b*1024 + f)

  __shared__ __bf16 lb[2][64][32];         // double-buffered weight tile
  v8f acc[4] = {};

  // per-thread slice of the cooperative tile copy: n = t/4, 8 k at (t%4)*8
  int t = threadIdx.x, nn = t >> 2, kc = (t & 3) * 8;
  const __bf16* wrow = &W[(n0 + nn) * HIDDEN + kc];
  __bf16* ldst[2] = { &lb[0][nn][kc], &lb[1][nn][kc] };

  async_copy16(wrow, ldst[0]);                       // tile 0
  const __bf16* arow = X + rw * HIDDEN;
  v16bf a = load_a_bf16(arow, HIDDEN, lane);         // A tile 0

  for (int i = 0; i < HIDDEN / 32; i++) {
    int buf = i & 1;
    WAIT_ASYNC(0);          // my async copy for tile i has landed in LDS
    __syncthreads();        // everyone's slice landed

    if (i + 1 < HIDDEN / 32)
      async_copy16(wrow + (i + 1) * 32, ldst[buf ^ 1]);  // prefetch tile i+1

    v16bf a_next = a;
    if (i + 1 < HIDDEN / 32)
      a_next = load_a_bf16(arow + (i + 1) * 32, HIDDEN, lane);

    const __bf16* lbuf = &lb[buf][0][0];
    // load all four B fragments first, then burst the WMMAs
    v16bf bf0 = load_b_bf16(lbuf + 0 * 512, 32, lane);
    v16bf bf1 = load_b_bf16(lbuf + 1 * 512, 32, lane);
    v16bf bf2 = load_b_bf16(lbuf + 2 * 512, 32, lane);
    v16bf bf3 = load_b_bf16(lbuf + 3 * 512, 32, lane);
    acc[0] = wmma_bf16(a, bf0, acc[0]);
    acc[1] = wmma_bf16(a, bf1, acc[1]);
    acc[2] = wmma_bf16(a, bf2, acc[2]);
    acc[3] = wmma_bf16(a, bf3, acc[3]);

    a = a_next;
    __syncthreads();        // all waves done reading buf before it is reused
  }

  int b = rw >> 10, fpos = rw & 1023;
  int bh = b * NHEAD + h;
  if (z < 2) {
    __bf16* out = (z == 0) ? Qb : Kb;      // [BH, seq, 64]
#pragma unroll
    for (int j = 0; j < 4; j++) {
      int d = j * 16 + (lane & 15);
      float bb = bias[n0 + d];
#pragma unroll
      for (int i = 0; i < 8; i++) {
        int row = fpos + i + 8 * (lane >> 4);  // C layout: M = i + 8*(lane/16)
        out[(bh * SEQ + row) * DH + d] = (__bf16)(acc[j][i] + bb);
      }
    }
  } else {
    // V stored transposed: Vt[bh][d][t]; consecutive acc elems -> consecutive t
#pragma unroll
    for (int j = 0; j < 4; j++) {
      int d = j * 16 + (lane & 15);
      float bb = bias[n0 + d];
      v8bf pk;
#pragma unroll
      for (int i = 0; i < 8; i++) pk[i] = (__bf16)(acc[j][i] + bb);
      int tbase = fpos + 8 * (lane >> 4);
      *(v8bf*)&Vt[(bh * DH + d) * SEQ + tbase] = pk;
    }
  }
}

// ---------------------------------------------------------------------------
// K2: scores + mask + softmax, fused. grid = (16 f-chunks, 64 bh), block = 128
// (4 waves). Each wave owns a 16x1024 score strip staged in LDS (256 KB total).
// K-tile B fragments are software-pipelined one t-tile ahead.
__global__ __launch_bounds__(128) void scores_softmax_kernel(
    const __bf16* __restrict__ Qb, const __bf16* __restrict__ Kb,
    const float* __restrict__ mask, float* __restrict__ probs) {
  __shared__ float sc[4][16][1024];
  int wave = threadIdx.x >> 5, lane = threadIdx.x & 31;
  int bh = blockIdx.y, b = bh >> 4;
  int f0 = blockIdx.x * 64 + wave * 16;

  const __bf16* Qbase = Qb + (bh * SEQ + f0) * DH;
  v16bf a0 = load_a_bf16(Qbase, DH, lane);        // d = 0..31
  v16bf a1 = load_a_bf16(Qbase + 32, DH, lane);   // d = 32..63

  const __bf16* Kbh = Kb + (size_t)bh * SEQ * DH;
  v16bf b0 = load_b_bf16(Kbh, DH, lane);
  v16bf b1 = load_b_bf16(Kbh + 32, DH, lane);

  for (int j = 0; j < 64; j++) {
    v16bf bn0 = b0, bn1 = b1;
    if (j + 1 < 64) {
      const __bf16* Kn = Kbh + (j + 1) * 16 * DH;
      bn0 = load_b_bf16(Kn, DH, lane);
      bn1 = load_b_bf16(Kn + 32, DH, lane);
    }
    v8f c = {};
    c = wmma_bf16(a0, b0, c);
    c = wmma_bf16(a1, b1, c);

    int tcol = j * 16 + (lane & 15);
#pragma unroll
    for (int i = 0; i < 8; i++) {
      int M = i + 8 * (lane >> 4);
      int f = f0 + M;
      float madd = (1.0f - mask[(size_t)(b * SEQ + f) * SEQ + tcol]) * -10000.0f;
      sc[wave][M][tcol] = c[i] * 0.125f + madd;   // 1/sqrt(64)
    }
    b0 = bn0; b1 = bn1;
  }
  __syncthreads();

  // softmax: 2 lanes per row, 512 columns each; combine via xor-16 shuffle.
  int row = lane & 15, half = lane >> 4;
  float* rp = &sc[wave][row][half * 512];
  float mx = -3.0e38f;
  for (int t = 0; t < 512; t++) mx = fmaxf(mx, rp[t]);
  mx = fmaxf(mx, __shfl_xor(mx, 16, 32));
  float sum = 0.f;
  for (int t = 0; t < 512; t++) { float e = __expf(rp[t] - mx); rp[t] = e; sum += e; }
  sum += __shfl_xor(sum, 16, 32);
  float inv = 1.0f / sum;

  float* op = probs + (size_t)(bh * SEQ + f0 + row) * SEQ + half * 512;
  for (int t = 0; t < 512; t += 4) {
    float4 v = *(float4*)&rp[t];
    v.x *= inv; v.y *= inv; v.z *= inv; v.w *= inv;
    *(float4*)&op[t] = v;
  }
}

// ---------------------------------------------------------------------------
// K3: ctx = P.V. grid = (8 f-tiles, 64 bh), block = 256 (8 waves).
// Wave: 16 f-rows x 64 d-cols; K = T = 1024 in steps of 32; A pipelined,
// B fragments batch-loaded before the WMMA burst.
__global__ __launch_bounds__(256) void ctx_gemm_kernel(
    const float* __restrict__ probs, const __bf16* __restrict__ Vt,
    float* __restrict__ ctx) {
  int wave = threadIdx.x >> 5, lane = threadIdx.x & 31;
  int bh = blockIdx.y, b = bh >> 4, h = bh & 15;
  int f0 = blockIdx.x * 128 + wave * 16;

  const float* Pbase = probs + (size_t)(bh * SEQ + f0) * SEQ;
  const __bf16* Vbase = Vt + (size_t)bh * DH * SEQ;   // [d][t]
  v8f acc[4] = {};

  v16bf a = load_a_f32(Pbase, SEQ, lane);
  for (int k0 = 0; k0 < SEQ; k0 += 32) {
    v16bf a_next = a;
    if (k0 + 32 < SEQ) a_next = load_a_f32(Pbase + k0 + 32, SEQ, lane);

    v16bf bf0 = load_b_bf16(Vbase + (size_t)(0 * 16) * SEQ + k0, SEQ, lane);
    v16bf bf1 = load_b_bf16(Vbase + (size_t)(1 * 16) * SEQ + k0, SEQ, lane);
    v16bf bf2 = load_b_bf16(Vbase + (size_t)(2 * 16) * SEQ + k0, SEQ, lane);
    v16bf bf3 = load_b_bf16(Vbase + (size_t)(3 * 16) * SEQ + k0, SEQ, lane);
    acc[0] = wmma_bf16(a, bf0, acc[0]);
    acc[1] = wmma_bf16(a, bf1, acc[1]);
    acc[2] = wmma_bf16(a, bf2, acc[2]);
    acc[3] = wmma_bf16(a, bf3, acc[3]);

    a = a_next;
  }

#pragma unroll
  for (int j = 0; j < 4; j++) {
    int n = j * 16 + (lane & 15);
#pragma unroll
    for (int i = 0; i < 8; i++) {
      int f = f0 + i + 8 * (lane >> 4);
      ctx[(size_t)(b * SEQ + f) * HIDDEN + h * DH + n] = acc[j][i];
    }
  }
}

// ---------------------------------------------------------------------------
extern "C" void kernel_launch(void* const* d_in, const int* in_sizes, int n_in,
                              void* d_out, int out_size, void* d_ws, size_t ws_size,
                              hipStream_t stream) {
  const float* from = (const float*)d_in[0];
  const float* to   = (const float*)d_in[1];
  const float* mask = (const float*)d_in[2];
  const float* Wq   = (const float*)d_in[3];
  const float* bq   = (const float*)d_in[4];
  const float* Wk   = (const float*)d_in[5];
  const float* bk   = (const float*)d_in[6];
  const float* Wv   = (const float*)d_in[7];
  const float* bv   = (const float*)d_in[8];

  char* ws = (char*)d_ws;
  __bf16* Xf  = (__bf16*)(ws);                       //  8 MB [4096,1024]
  __bf16* Xt  = (__bf16*)(ws + (size_t)(8u  << 20)); //  8 MB
  __bf16* Wqt = (__bf16*)(ws + (size_t)(16u << 20)); //  2 MB [N,K]
  __bf16* Wkt = (__bf16*)(ws + (size_t)(18u << 20));
  __bf16* Wvt = (__bf16*)(ws + (size_t)(20u << 20));
  __bf16* Qb  = (__bf16*)(ws + (size_t)(22u << 20)); //  8 MB [BH,F,64]
  __bf16* Kb  = (__bf16*)(ws + (size_t)(30u << 20)); //  8 MB [BH,T,64]
  __bf16* Vt  = (__bf16*)(ws + (size_t)(38u << 20)); //  8 MB [BH,64,T]  (46 MB total)

  float* ctx   = (float*)d_out;                                  // [B,F,1024]
  float* probs = (float*)d_out + (size_t)BATCH * SEQ * HIDDEN;   // [B,H,F,T]

  cvt_kernel<<<4096, 256, 0, stream>>>(from, Xf);
  cvt_kernel<<<4096, 256, 0, stream>>>(to, Xt);
  transpose_cvt_kernel<<<dim3(32, 32), 256, 0, stream>>>(Wq, Wqt);
  transpose_cvt_kernel<<<dim3(32, 32), 256, 0, stream>>>(Wk, Wkt);
  transpose_cvt_kernel<<<dim3(32, 32), 256, 0, stream>>>(Wv, Wvt);

  qkv_gemm_kernel<<<dim3(16, 32, 3), 256, 0, stream>>>(
      Xf, Xt, Wqt, Wkt, Wvt, bq, bk, bv, Qb, Kb, Vt);

  scores_softmax_kernel<<<dim3(16, 64), 128, 0, stream>>>(Qb, Kb, mask, probs);

  ctx_gemm_kernel<<<dim3(8, 64), 256, 0, stream>>>(probs, Vt, ctx);
}
